// MFGNN_14894946583444
// MI455X (gfx1250) — compile-verified, matrix operand order checked by reading, not run on hardware
//
#include <hip/hip_runtime.h>

typedef __attribute__((ext_vector_type(16))) _Float16 v16h;
typedef __attribute__((ext_vector_type(8)))  float    v8f;

#define N_NODES 100000
#define N_EDGES 1600000

// ---------------------------------------------------------------- zero
__global__ void mfgnn_zero(float* __restrict__ p, int n4) {
    int i = blockIdx.x * blockDim.x + threadIdx.x;
    if (i < n4) ((float4*)p)[i] = make_float4(0.f, 0.f, 0.f, 0.f);
}

// ---------------------------------------------------------------- concat [x | add_x] -> h0
__global__ void mfgnn_concat(const float* __restrict__ x, const float* __restrict__ ax,
                             float* __restrict__ h) {
    int i = blockIdx.x * blockDim.x + threadIdx.x;   // one float4 per thread
    int n  = i >> 5;                                 // 32 float4 per 128-wide row
    int c4 = i & 31;
    if (n >= N_NODES) return;
    const float* srcm = (c4 < 16) ? (x + (long long)n * 64) : (ax + (long long)n * 64);
    float4 v = ((const float4*)srcm)[c4 & 15];
    ((float4*)(h + (long long)n * 128))[c4] = v;
}

// ---------------------------------------------------------------- edge scatter-add
// 1 wave per edge; lane l handles dims [4l, 4l+4). agg[dst] += h[src].
__global__ void mfgnn_scatter(const float* __restrict__ h,
                              const long long* __restrict__ src,
                              const long long* __restrict__ dst,
                              float* __restrict__ agg) {
    long long tid = (long long)blockIdx.x * blockDim.x + threadIdx.x;
    int e    = (int)(tid >> 5);
    int lane = (int)(tid & 31);
    if (e >= N_EDGES) return;
    int s = (int)src[e];
    int d = (int)dst[e];
    float4 v = ((const float4*)(h + (long long)s * 128))[lane];
    float* o = agg + (long long)d * 128 + lane * 4;
    atomicAdd(o + 0, v.x);
    atomicAdd(o + 1, v.y);
    atomicAdd(o + 2, v.z);
    atomicAdd(o + 3, v.w);
}

// ---------------------------------------------------------------- fused GraphConv GEMM
// out[16 rows x dout] = [agg | h] (16x256) @ [wrel; wroot] (256 x dout) + b (+relu)(+residual)
// Block: (dout/16) waves; wave w owns output col-tile w. K-loop: 8 chunks of 32.
// Weights staged once per block into LDS in exact B-fragment layout (f16).
__global__ void mfgnn_gemm(const float* __restrict__ agg, const float* __restrict__ h,
                           const float* __restrict__ wrel, const float* __restrict__ brel,
                           const float* __restrict__ wroot, float* __restrict__ out,
                           const float* __restrict__ residual, int dout, int doRelu) {
    extern __shared__ _Float16 ldsB[];   // [dout/16][8 kchunks][32 lanes][16 halves]

    const int tid = threadIdx.x;
    const int nThreads = blockDim.x;

    // ---- stage combined weight matrix (256 x dout) into B-fragment layout
    const int total = 256 * dout;
    for (int idx = tid; idx < total; idx += nThreads) {
        int k = idx / dout;
        int n = idx - k * dout;
        float w = (k < 128) ? wrel[k * dout + n] : wroot[(k - 128) * dout + n];
        int tile = n >> 4, col = n & 15;
        int kc = k >> 5, kin = k & 31;
        // B fragment layout (mirror of 16-bit A layout): lane = col + 16*(bit3 of kin),
        // element jj = (kin&7) + 8*(bit4 of kin)
        int lane = col + ((kin >> 3) & 1) * 16;
        int jj   = (kin & 7) + ((kin >> 4) & 1) * 8;
        ldsB[((tile * 8 + kc) * 32 + lane) * 16 + jj] = (_Float16)w;
    }
    __syncthreads();

    const int wave = tid >> 5;
    const int lane = tid & 31;
    const int arow = blockIdx.x * 16 + (lane & 15);   // A-fragment row for this lane
    const int khalf = (lane >> 4) * 8;                // K sub-base within a 32-chunk

    const float* hrow = h   + (long long)arow * 128;
    const float* grow = agg + (long long)arow * 128;

    v8f c = {};
#pragma unroll
    for (int kc = 0; kc < 8; ++kc) {
        const int kg = kc * 32;
        const float* srcm = (kg < 128) ? grow : hrow;   // [agg | h] virtual concat
        const int colb = (kg & 127) + khalf;
        float4 p0 = *(const float4*)(srcm + colb);
        float4 p1 = *(const float4*)(srcm + colb + 4);
        float4 p2 = *(const float4*)(srcm + colb + 16);
        float4 p3 = *(const float4*)(srcm + colb + 20);
        v16h a;
        a[0]  = (_Float16)p0.x; a[1]  = (_Float16)p0.y; a[2]  = (_Float16)p0.z; a[3]  = (_Float16)p0.w;
        a[4]  = (_Float16)p1.x; a[5]  = (_Float16)p1.y; a[6]  = (_Float16)p1.z; a[7]  = (_Float16)p1.w;
        a[8]  = (_Float16)p2.x; a[9]  = (_Float16)p2.y; a[10] = (_Float16)p2.z; a[11] = (_Float16)p2.w;
        a[12] = (_Float16)p3.x; a[13] = (_Float16)p3.y; a[14] = (_Float16)p3.z; a[15] = (_Float16)p3.w;

        v16h b = *(const v16h*)(ldsB + ((wave * 8 + kc) * 32 + lane) * 16);

        c = __builtin_amdgcn_wmma_f32_16x16x32_f16(
                /*neg_a=*/false, a, /*neg_b=*/false, b,
                /*c_mod=*/(short)0, c, /*reuse_a=*/false, /*reuse_b=*/false);
    }

    // ---- epilogue: bias, relu, optional residual; C/D layout: VGPR i -> M = 8*(lane>=16)+i
    const int ncol = wave * 16 + (lane & 15);
    const float bias = brel[ncol];
    const int mbase = blockIdx.x * 16 + (lane >> 4) * 8;
#pragma unroll
    for (int i = 0; i < 8; ++i) {
        const long long m = mbase + i;
        float v = c[i] + bias;
        if (doRelu) v = fmaxf(v, 0.f);
        if (residual) v += residual[m * dout + ncol];
        out[m * dout + ncol] = v;
    }
}

// ---------------------------------------------------------------- launcher
extern "C" void kernel_launch(void* const* d_in, const int* in_sizes, int n_in,
                              void* d_out, int out_size, void* d_ws, size_t ws_size,
                              hipStream_t stream) {
    const float*     x      = (const float*)d_in[0];
    const float*     ax     = (const float*)d_in[1];
    const long long* ei     = (const long long*)d_in[2];   // int64 [2, E]
    const float*     w1rel  = (const float*)d_in[3];
    const float*     b1rel  = (const float*)d_in[4];
    const float*     w1root = (const float*)d_in[5];
    const float*     w2rel  = (const float*)d_in[6];
    const float*     b2rel  = (const float*)d_in[7];
    const float*     w2root = (const float*)d_in[8];
    const float*     w3rel  = (const float*)d_in[9];
    const float*     b3rel  = (const float*)d_in[10];
    const float*     w3root = (const float*)d_in[11];
    float*           outp   = (float*)d_out;

    const long long* src = ei;
    const long long* dst = ei + N_EDGES;

    float* hA  = (float*)d_ws;
    float* agg = hA  + (size_t)N_NODES * 128;
    float* hB  = agg + (size_t)N_NODES * 128;

    const int n4 = N_NODES * 128 / 4;                       // 3.2M float4
    const int zeroBlocks   = (n4 + 255) / 256;
    const int concatBlocks = (N_NODES * 32 + 255) / 256;
    const long long sThreads = (long long)N_EDGES * 32;
    const int scatterBlocks = (int)((sThreads + 255) / 256); // 200k blocks
    const int gemmGrid = N_NODES / 16;                       // 6250 exactly
    const size_t lds128 = (size_t)128 * 256 * sizeof(_Float16); // 64 KB
    const size_t lds64  = (size_t)64  * 256 * sizeof(_Float16); // 32 KB

    // h0 = [x | additional_x]
    mfgnn_concat<<<concatBlocks, 256, 0, stream>>>(x, ax, hA);

    // ---- layer 1: hB = relu(GC(hA))
    mfgnn_zero<<<zeroBlocks, 256, 0, stream>>>(agg, n4);
    mfgnn_scatter<<<scatterBlocks, 256, 0, stream>>>(hA, src, dst, agg);
    mfgnn_gemm<<<gemmGrid, 256, lds128, stream>>>(agg, hA, w1rel, b1rel, w1root,
                                                  hB, nullptr, 128, 1);

    // ---- layer 2: hA = relu(GC(hB))
    mfgnn_zero<<<zeroBlocks, 256, 0, stream>>>(agg, n4);
    mfgnn_scatter<<<scatterBlocks, 256, 0, stream>>>(hB, src, dst, agg);
    mfgnn_gemm<<<gemmGrid, 256, lds128, stream>>>(agg, hB, w2rel, b2rel, w2root,
                                                  hA, nullptr, 128, 1);

    // ---- layer 3: out = GC(hA) + additional_x   (dout = 64 -> 4 waves)
    mfgnn_zero<<<zeroBlocks, 256, 0, stream>>>(agg, n4);
    mfgnn_scatter<<<scatterBlocks, 256, 0, stream>>>(hA, src, dst, agg);
    mfgnn_gemm<<<gemmGrid, 128, lds64, stream>>>(agg, hA, w3rel, b3rel, w3root,
                                                 outp, ax, 64, 0);
}
